// OptimizedPackedBitLinear_57939108823184
// MI455X (gfx1250) — matrix operand order repaired and checked by reading, not compiled, not run on hardware
//
#include <hip/hip_runtime.h>
#include <cstdint>

typedef __bf16 bf16;
typedef __attribute__((ext_vector_type(16))) __bf16 v16bf;
typedef __attribute__((ext_vector_type(8)))  __bf16 v8bf;
typedef __attribute__((ext_vector_type(4)))  __bf16 v4bf;
typedef __attribute__((ext_vector_type(8)))  float  v8f;

#define B_DIM   8192
#define IN_DIM  4096
#define OUT_DIM 4096
#define EPSV    1e-5f

// ---------------------------------------------------------------------------
// Pass 1: per-row absmax scale + bf16 quantization of x/xs.
// One 256-thread block per row; each thread handles 4 float4 (16 elements).
// ---------------------------------------------------------------------------
__global__ __launch_bounds__(256) void bitlinear_scale_quant(
    const float* __restrict__ x, bf16* __restrict__ xq,
    float* __restrict__ srow)
{
    const int row = blockIdx.x;
    const int tid = threadIdx.x;
    const float4* xr = (const float4*)(x + (size_t)row * IN_DIM);

    float4 v[4];
    float m = 0.0f;
#pragma unroll
    for (int t = 0; t < 4; ++t) {
        float4 f = xr[tid + t * 256];
        v[t] = f;
        m = fmaxf(m, fmaxf(fmaxf(fabsf(f.x), fabsf(f.y)),
                           fmaxf(fabsf(f.z), fabsf(f.w))));
    }

    __shared__ float red[256];
    red[tid] = m;
    __syncthreads();
#pragma unroll
    for (int s = 128; s > 0; s >>= 1) {
        if (tid < s) red[tid] = fmaxf(red[tid], red[tid + s]);
        __syncthreads();
    }
    const float sv = fmaxf(red[0], EPSV);
    if (tid == 0) srow[row] = sv;
    const float inv = 1.0f / sv;

    v4bf* oq = (v4bf*)(xq + (size_t)row * IN_DIM);
#pragma unroll
    for (int t = 0; t < 4; ++t) {
        float4 f = v[t];
        v4bf o;
        o[0] = (bf16)(f.x * inv);
        o[1] = (bf16)(f.y * inv);
        o[2] = (bf16)(f.z * inv);
        o[3] = (bf16)(f.w * inv);
        oq[tid + t * 256] = o;
    }
}

// ---------------------------------------------------------------------------
// Pass 2: decode ternary weights (pos - neg) into bf16 W[OUT, IN] (row-major).
// 8 elements per thread: two u32 mask loads -> one 16B bf16 store.
// ---------------------------------------------------------------------------
__global__ __launch_bounds__(256) void bitlinear_make_w(
    const unsigned char* __restrict__ pos, const unsigned char* __restrict__ neg,
    bf16* __restrict__ wq)
{
    const size_t i = ((size_t)blockIdx.x * 256 + threadIdx.x) * 8;
    uint2 p = *(const uint2*)(pos + i);
    uint2 q = *(const uint2*)(neg + i);
    v8bf o;
#pragma unroll
    for (int b = 0; b < 4; ++b) {
        o[b]     = (bf16)((float)((p.x >> (8*b)) & 0xff) - (float)((q.x >> (8*b)) & 0xff));
        o[b + 4] = (bf16)((float)((p.y >> (8*b)) & 0xff) - (float)((q.y >> (8*b)) & 0xff));
    }
    *(v8bf*)(wq + i) = o;
}

// ---------------------------------------------------------------------------
// Main GEMM: out[B, OUT] = (xq @ Wq^T) * scale * srow[row]
// Block tile 128x256 (8 waves, 2x4), wave tile 64x64 = 16 WMMA accumulators.
// v_wmma_f32_16x16x32_bf16, fragments loaded straight from global (W is
// L2-resident at 32 MB; grid ordering reuses each A stripe across N-tiles).
// ---------------------------------------------------------------------------
__global__ __launch_bounds__(256, 1) void bitlinear_wmma_gemm(
    const bf16* __restrict__ Aq, const bf16* __restrict__ Wq,
    const float* __restrict__ srow, const float* __restrict__ scale_ptr,
    float* __restrict__ out)
{
    const int lane = threadIdx.x & 31;
    const int wave = threadIdx.x >> 5;
    const int wm = wave >> 2;          // 0..1  (M direction)
    const int wn = wave & 3;           // 0..3  (N direction)
    const int half = lane >> 4;        // 0/1
    const int l16  = lane & 15;
    const int m0 = blockIdx.y * 128 + wm * 64;
    const int n0 = blockIdx.x * 256 + wn * 64;

    // A fragment (16x32 bf16): lane holds row (l16), K chunks [half*8, +8) and +16.
    const bf16* aPtr[4];
#pragma unroll
    for (int i = 0; i < 4; ++i)
        aPtr[i] = Aq + (size_t)(m0 + 16*i + l16) * IN_DIM + half * 8;

    // B fragment (32x16 bf16): lane holds column (l16) = W row, K chunks [half*8, +8) and +16.
    const bf16* bPtr[4];
#pragma unroll
    for (int j = 0; j < 4; ++j)
        bPtr[j] = Wq + (size_t)(n0 + 16*j + l16) * IN_DIM + half * 8;

    v8f acc[4][4] = {};

#pragma unroll 2
    for (int k = 0; k < IN_DIM; k += 32) {
        v16bf a[4], b[4];
#pragma unroll
        for (int i = 0; i < 4; ++i) {
            v8bf lo = *(const v8bf*)(aPtr[i] + k);
            v8bf hi = *(const v8bf*)(aPtr[i] + k + 16);
            a[i] = __builtin_shufflevector(lo, hi,
                     0,1,2,3,4,5,6,7,8,9,10,11,12,13,14,15);
        }
#pragma unroll
        for (int j = 0; j < 4; ++j) {
            v8bf lo = *(const v8bf*)(bPtr[j] + k);
            v8bf hi = *(const v8bf*)(bPtr[j] + k + 16);
            b[j] = __builtin_shufflevector(lo, hi,
                     0,1,2,3,4,5,6,7,8,9,10,11,12,13,14,15);
        }
#pragma unroll
        for (int i = 0; i < 4; ++i)
#pragma unroll
            for (int j = 0; j < 4; ++j)
                acc[i][j] = __builtin_amdgcn_wmma_f32_16x16x32_bf16(
                    false, a[i], false, b[j], (short)0, acc[i][j], false, false);
    }

    // Epilogue: D VGPR r <-> row (half*8 + r), column l16.
    const float sc = *scale_ptr;
#pragma unroll
    for (int i = 0; i < 4; ++i) {
        const int mbase = m0 + 16*i + half * 8;
        float f[8];
#pragma unroll
        for (int r = 0; r < 8; ++r) f[r] = sc * srow[mbase + r];
#pragma unroll
        for (int j = 0; j < 4; ++j) {
            const int col = n0 + 16*j + l16;
#pragma unroll
            for (int r = 0; r < 8; ++r)
                out[(size_t)(mbase + r) * OUT_DIM + col] = acc[i][j][r] * f[r];
        }
    }
}

// ---------------------------------------------------------------------------
extern "C" void kernel_launch(void* const* d_in, const int* in_sizes, int n_in,
                              void* d_out, int out_size, void* d_ws, size_t ws_size,
                              hipStream_t stream)
{
    const float*         x    = (const float*)d_in[0];
    const unsigned char* pos  = (const unsigned char*)d_in[1];  // jnp.bool_ = 1 byte
    const unsigned char* neg  = (const unsigned char*)d_in[2];
    const float*         scal = (const float*)d_in[3];
    float*               out  = (float*)d_out;

    // Workspace layout: [srow: 8192 f32][xq: B*IN bf16 = 64MB][wq: OUT*IN bf16 = 32MB]
    char* ws = (char*)d_ws;
    float* srow = (float*)ws;
    bf16*  xq   = (bf16*)(ws + 64 * 1024);
    bf16*  wq   = (bf16*)(ws + 64 * 1024 + (size_t)B_DIM * IN_DIM * sizeof(bf16));

    bitlinear_scale_quant<<<B_DIM, 256, 0, stream>>>(x, xq, srow);

    const size_t wElems = (size_t)OUT_DIM * IN_DIM;          // 16,777,216
    bitlinear_make_w<<<(unsigned)(wElems / (8 * 256)), 256, 0, stream>>>(pos, neg, wq);

    dim3 grid(OUT_DIM / 256, B_DIM / 128);                   // (16, 64): N-tiles inner for L2 reuse
    bitlinear_wmma_gemm<<<grid, 256, 0, stream>>>(xq, wq, srow, scal, out);
}
